// NetMamba_49469433315666
// MI455X (gfx1250) — compile-verified, matrix operand order checked by reading, not run on hardware
//
#include <hip/hip_runtime.h>
#include <hip/hip_bf16.h>

#define D_MODELC 192
#define D_INNERC 384
#define D_STATEC 16
#define DT_RANKC 12
#define D_CONVC  4
#define DEPTHC   4
#define N_PATCHC 400
#define L_SEQC   401
#define BATCHC   128
#define M_ROWSC  (BATCHC * L_SEQC)   /* 51328, multiple of 32 */
#define XP_PADC  64                  /* 44 padded to 2 N-groups of 32 */
#define EPSC     1e-5f

typedef __attribute__((ext_vector_type(16))) __bf16 v16bf;
typedef __attribute__((ext_vector_type(8)))  __bf16 v8bf;
typedef __attribute__((ext_vector_type(8)))  float  v8f;

// Load one 16x32 bf16 WMMA fragment row-slice for this lane:
// two contiguous 16B chunks at p and p+16 (K phase handled by caller via +hi8).
__device__ __forceinline__ v16bf ldfrag(const __bf16* p) {
  v8bf lo = *(const v8bf*)(p);
  v8bf hi = *(const v8bf*)(p + 16);
  return __builtin_shufflevector(lo, hi, 0, 1, 2, 3, 4, 5, 6, 7,
                                 8, 9, 10, 11, 12, 13, 14, 15);
}

__device__ __forceinline__ v8f wmma_bf16(v16bf a, v16bf b, v8f c) {
  return __builtin_amdgcn_wmma_f32_16x16x32_bf16(false, a, false, b,
                                                 (short)0, c, false, false);
}

// ---------------------------------------------------------------------------
// Patch embedding + positional embedding + cls token -> hidden [B, L, 192] f32
// ---------------------------------------------------------------------------
__global__ void patch_embed_kernel(const float* __restrict__ imgs,
                                   const float* __restrict__ pe_w,
                                   const float* __restrict__ pe_b,
                                   const float* __restrict__ cls,
                                   const float* __restrict__ pos,
                                   float* __restrict__ hidden) {
  int idx = blockIdx.x * blockDim.x + threadIdx.x;
  int total = BATCHC * L_SEQC * D_MODELC;
  if (idx >= total) return;
  int c = idx % D_MODELC;
  int l = (idx / D_MODELC) % L_SEQC;
  int b = idx / (D_MODELC * L_SEQC);
  float v;
  if (l < N_PATCHC) {
    const float* p = imgs + b * 1600 + l * 4;
    v = pe_b[c] + pos[l * D_MODELC + c]
      + pe_w[c * 4 + 0] * p[0] + pe_w[c * 4 + 1] * p[1]
      + pe_w[c * 4 + 2] * p[2] + pe_w[c * 4 + 3] * p[3];
  } else {
    v = cls[c] + pos[N_PATCHC * D_MODELC + c];
  }
  hidden[idx] = v;
}

// ---------------------------------------------------------------------------
// residual = (first ? hidden : hidden + residual); out_bf = bf16(rms(residual)*w)
// one wave32 per row of 192; wave shuffle reduction
// ---------------------------------------------------------------------------
__global__ __launch_bounds__(32) void rms_norm_kernel(const float* __restrict__ hidden,
                                                      float* __restrict__ residual,
                                                      const float* __restrict__ w,
                                                      __bf16* __restrict__ out_bf,
                                                      int first) {
  int row = blockIdx.x;
  int lane = threadIdx.x;
  const float* h = hidden + (size_t)row * D_MODELC;
  float* r = residual + (size_t)row * D_MODELC;
  float vals[6];
  float ss = 0.f;
#pragma unroll
  for (int i = 0; i < 6; ++i) {
    int c = lane + i * 32;
    float x = h[c];
    if (!first) x += r[c];
    r[c] = x;
    vals[i] = x;
    ss += x * x;
  }
#pragma unroll
  for (int off = 16; off > 0; off >>= 1) ss += __shfl_xor(ss, off, 32);
  float scale = rsqrtf(ss * (1.0f / D_MODELC) + EPSC);
#pragma unroll
  for (int i = 0; i < 6; ++i) {
    int c = lane + i * 32;
    out_bf[(size_t)row * D_MODELC + c] = (__bf16)(vals[i] * scale * w[c]);
  }
}

// ---------------------------------------------------------------------------
// f32 -> bf16 elementwise (weights / activations)
// ---------------------------------------------------------------------------
__global__ void cvt_bf16_kernel(const float* __restrict__ in,
                                __bf16* __restrict__ out, int n) {
  int i = blockIdx.x * blockDim.x + threadIdx.x;
  if (i < n) out[i] = (__bf16)in[i];
}

// x_proj weight [44,384] -> bf16 padded to [64,384] (zero rows 44..63)
__global__ void cvt_xp_pad_kernel(const float* __restrict__ w,
                                  __bf16* __restrict__ out) {
  int i = blockIdx.x * blockDim.x + threadIdx.x;
  if (i >= XP_PADC * D_INNERC) return;
  int n = i / D_INNERC, k = i % D_INNERC;
  out[i] = (n < DT_RANKC + 2 * D_STATEC) ? (__bf16)w[n * D_INNERC + k] : (__bf16)0.f;
}

// ---------------------------------------------------------------------------
// C[M,N] = A[M,K] * W[N,K]^T  via v_wmma_f32_16x16x32_bf16
// Register-blocked: one wave owns a 32x32 output tile = 4 accumulators.
// Per K-step: 2 A-frags + 2 B-frags feed 4 independent WMMAs (16 FLOP/byte),
// with explicit double-buffering so fragment loads for K+32 are in flight
// while the current 4 WMMAs execute. Wave-uniform guard keeps EXEC all-1s.
// Requires M%32==0, N%32==0, K%32==0.
// ---------------------------------------------------------------------------
__global__ __launch_bounds__(256) void wmma_gemm_bf16_kernel(
    const __bf16* __restrict__ A, const __bf16* __restrict__ W,
    float* __restrict__ C, int M, int N, int K) {
  int wavesPerBlock = blockDim.x >> 5;
  int wid = blockIdx.x * wavesPerBlock + ((int)threadIdx.x >> 5);
  int ntn = N >> 5;                         // 32-col groups
  int ntiles = (M >> 5) * ntn;
  if (wid >= ntiles) return;                // wave-uniform
  int tm = (wid / ntn) << 5;
  int tn = (wid % ntn) << 5;
  int lane = threadIdx.x & 31;
  int row  = lane & 15;
  int hi8  = (lane >> 4) << 3;              // K phase: 0 or 8
  const __bf16* a0p = A + (size_t)(tm + row) * K + hi8;
  const __bf16* a1p = a0p + (size_t)16 * K;
  const __bf16* b0p = W + (size_t)(tn + row) * K + hi8;
  const __bf16* b1p = b0p + (size_t)16 * K;

  v8f acc00 = {}, acc01 = {}, acc10 = {}, acc11 = {};

  // preload K-block 0
  v16bf a0 = ldfrag(a0p), a1 = ldfrag(a1p);
  v16bf b0 = ldfrag(b0p), b1 = ldfrag(b1p);

  for (int k0 = 0; k0 < K; k0 += 32) {
    bool more = (k0 + 32) < K;
    v16bf na0 = a0, na1 = a1, nb0 = b0, nb1 = b1;
    if (more) {
      // issue next K-block fragment loads before this block's WMMAs
      __builtin_prefetch(a0p + k0 + 64, 0, 1);   // global_prefetch_b8 (L2 path)
      __builtin_prefetch(b0p + k0 + 64, 0, 1);
      na0 = ldfrag(a0p + k0 + 32);
      na1 = ldfrag(a1p + k0 + 32);
      nb0 = ldfrag(b0p + k0 + 32);
      nb1 = ldfrag(b1p + k0 + 32);
    }
    acc00 = wmma_bf16(a0, b0, acc00);
    acc01 = wmma_bf16(a0, b1, acc01);
    acc10 = wmma_bf16(a1, b0, acc10);
    acc11 = wmma_bf16(a1, b1, acc11);
    a0 = na0; a1 = na1; b0 = nb0; b1 = nb1;
  }

  // C/D layout: lane(0..15): vgpr r -> (M=tile_m+r, N=tn+lane);
  //             lane+16:     vgpr r -> (M=tile_m+8+r, N=tn+lane)
  float* c0 = C + (size_t)(tm + hi8) * N + tn + row;
  float* c1 = C + (size_t)(tm + 16 + hi8) * N + tn + row;
#pragma unroll
  for (int r = 0; r < 8; ++r) {
    c0[(size_t)r * N]      = acc00[r];
    c0[(size_t)r * N + 16] = acc01[r];
    c1[(size_t)r * N]      = acc10[r];
    c1[(size_t)r * N + 16] = acc11[r];
  }
}

// ---------------------------------------------------------------------------
// depthwise causal conv (width 4) over sequence + SiLU; reads xi half of xz
// ---------------------------------------------------------------------------
__global__ void conv_silu_kernel(const float* __restrict__ xz,
                                 const float* __restrict__ cw,
                                 const float* __restrict__ cb,
                                 float* __restrict__ xc,
                                 __bf16* __restrict__ xc_bf) {
  int idx = blockIdx.x * blockDim.x + threadIdx.x;
  if (idx >= M_ROWSC * D_INNERC) return;
  int d = idx % D_INNERC;
  int m = idx / D_INNERC;
  int l = m % L_SEQC;
  int b = m / L_SEQC;
  float acc = cb[d];
#pragma unroll
  for (int k = 0; k < D_CONVC; ++k) {
    int ls = l + k - (D_CONVC - 1);
    if (ls >= 0)
      acc += cw[d * D_CONVC + k] *
             xz[(size_t)(b * L_SEQC + ls) * (2 * D_INNERC) + d];
  }
  float s = acc / (1.f + __expf(-acc));   // silu
  xc[idx] = s;
  xc_bf[idx] = (__bf16)s;
}

// dt = softplus(dbl[:, :12] @ dt_w^T + dt_b)   (K=12, VALU)
__global__ void dt_kernel(const float* __restrict__ dbl,
                          const float* __restrict__ dt_w,
                          const float* __restrict__ dt_b,
                          float* __restrict__ dt) {
  int idx = blockIdx.x * blockDim.x + threadIdx.x;
  if (idx >= M_ROWSC * D_INNERC) return;
  int d = idx % D_INNERC;
  int m = idx / D_INNERC;
  const float* r = dbl + (size_t)m * XP_PADC;
  float acc = dt_b[d];
#pragma unroll
  for (int j = 0; j < DT_RANKC; ++j) acc += dt_w[d * DT_RANKC + j] * r[j];
  dt[idx] = (acc > 20.f) ? acc : log1pf(__expf(acc));
}

// ---------------------------------------------------------------------------
// selective scan: 1 block per batch (384 threads = channel d), h[16] in regs,
// B/C vectors staged through LDS per timestep; fuses +D*x and *silu(z); bf16 out
// ---------------------------------------------------------------------------
__global__ __launch_bounds__(D_INNERC) void scan_kernel(
    const float* __restrict__ dbl, const float* __restrict__ dt,
    const float* __restrict__ xc,  const float* __restrict__ xz,
    const float* __restrict__ A_log, const float* __restrict__ Dp,
    __bf16* __restrict__ y_bf) {
  int b = blockIdx.x;
  int d = threadIdx.x;
  __shared__ float sB[D_STATEC];
  __shared__ float sC[D_STATEC];
  float A[D_STATEC], h[D_STATEC];
#pragma unroll
  for (int s = 0; s < D_STATEC; ++s) {
    A[s] = -__expf(A_log[d * D_STATEC + s]);
    h[s] = 0.f;
  }
  float Dd = Dp[d];
  for (int l = 0; l < L_SEQC; ++l) {
    size_t m = (size_t)b * L_SEQC + l;
    __syncthreads();
    if (d < 2 * D_STATEC) {
      float v = dbl[m * XP_PADC + DT_RANKC + d];
      if (d < D_STATEC) sB[d] = v; else sC[d - D_STATEC] = v;
    }
    __syncthreads();
    float dt_t = dt[m * D_INNERC + d];
    float x_t  = xc[m * D_INNERC + d];
    float y = 0.f;
#pragma unroll
    for (int s = 0; s < D_STATEC; ++s) {
      float dA = __expf(dt_t * A[s]);
      h[s] = h[s] * dA + dt_t * sB[s] * x_t;
      y += h[s] * sC[s];
    }
    y += Dd * x_t;
    float z = xz[m * (2 * D_INNERC) + D_INNERC + d];
    y *= z / (1.f + __expf(-z));          // * silu(z)
    y_bf[m * D_INNERC + d] = (__bf16)y;
  }
}

// final RMS norm on row l=400 only -> v_sem [B,192]
__global__ __launch_bounds__(32) void final_vsem_kernel(const float* __restrict__ hidden,
                                                        const float* __restrict__ residual,
                                                        const float* __restrict__ w,
                                                        float* __restrict__ v_sem) {
  int b = blockIdx.x;
  int lane = threadIdx.x;
  size_t row = (size_t)b * L_SEQC + N_PATCHC;
  float vals[6];
  float ss = 0.f;
#pragma unroll
  for (int i = 0; i < 6; ++i) {
    int c = lane + i * 32;
    float x = hidden[row * D_MODELC + c] + residual[row * D_MODELC + c];
    vals[i] = x;
    ss += x * x;
  }
#pragma unroll
  for (int off = 16; off > 0; off >>= 1) ss += __shfl_xor(ss, off, 32);
  float scale = rsqrtf(ss * (1.0f / D_MODELC) + EPSC);
#pragma unroll
  for (int i = 0; i < 6; ++i) {
    int c = lane + i * 32;
    v_sem[b * D_MODELC + c] = vals[i] * scale * w[c];
  }
}

// stat MLP: relu(stat@W1^T+b1)@W2^T+b2, stat=[pl|iat]; one block per batch row
__global__ __launch_bounds__(128) void stat_mlp_kernel(
    const float* __restrict__ pl, const float* __restrict__ iat,
    const float* __restrict__ w1, const float* __restrict__ b1,
    const float* __restrict__ w2, const float* __restrict__ b2,
    float* __restrict__ v_stat) {
  int b = blockIdx.x;
  int t = threadIdx.x;
  __shared__ float h1[128];
  float acc = b1[t];
  const float* wr = w1 + t * 200;
  for (int j = 0; j < 100; ++j) acc += wr[j] * pl[b * 100 + j];
  for (int j = 0; j < 100; ++j) acc += wr[100 + j] * iat[b * 100 + j];
  h1[t] = fmaxf(acc, 0.f);
  __syncthreads();
  float acc2 = b2[t];
  const float* w2r = w2 + t * 128;
  for (int j = 0; j < 128; ++j) acc2 += w2r[j] * h1[j];
  v_stat[b * 128 + t] = acc2;
}

// head: [v_sem|v_stat] @ head_w^T + head_b -> out [B,20]
__global__ __launch_bounds__(32) void head_kernel(const float* __restrict__ v_sem,
                                                  const float* __restrict__ v_stat,
                                                  const float* __restrict__ hw,
                                                  const float* __restrict__ hb,
                                                  float* __restrict__ out) {
  int b = blockIdx.x;
  int o = threadIdx.x;
  if (o >= 20) return;
  const float* wr = hw + o * (D_MODELC + 128);
  float acc = hb[o];
  for (int j = 0; j < D_MODELC; ++j) acc += wr[j] * v_sem[b * D_MODELC + j];
  for (int j = 0; j < 128; ++j) acc += wr[D_MODELC + j] * v_stat[b * 128 + j];
  out[b * 20 + o] = acc;
}

// ---------------------------------------------------------------------------
extern "C" void kernel_launch(void* const* d_in, const int* in_sizes, int n_in,
                              void* d_out, int out_size, void* d_ws, size_t ws_size,
                              hipStream_t stream) {
  (void)in_sizes; (void)n_in; (void)out_size; (void)ws_size;
  const float* imgs       = (const float*)d_in[0];
  const float* pl         = (const float*)d_in[1];
  const float* iat        = (const float*)d_in[2];
  const float* pe_w       = (const float*)d_in[3];
  const float* pe_b       = (const float*)d_in[4];
  const float* cls        = (const float*)d_in[5];
  const float* pos        = (const float*)d_in[6];
  const float* norm_ws    = (const float*)d_in[7];
  const float* in_proj_ws = (const float*)d_in[8];
  const float* conv_ws    = (const float*)d_in[9];
  const float* conv_bs    = (const float*)d_in[10];
  const float* x_proj_ws  = (const float*)d_in[11];
  const float* dt_proj_ws = (const float*)d_in[12];
  const float* dt_proj_bs = (const float*)d_in[13];
  const float* A_logs     = (const float*)d_in[14];
  const float* Ds         = (const float*)d_in[15];
  const float* out_proj_ws= (const float*)d_in[16];
  const float* norm_f_w   = (const float*)d_in[17];
  const float* mlp1_w     = (const float*)d_in[18];
  const float* mlp1_b     = (const float*)d_in[19];
  const float* mlp2_w     = (const float*)d_in[20];
  const float* mlp2_b     = (const float*)d_in[21];
  const float* head_w     = (const float*)d_in[22];
  const float* head_b     = (const float*)d_in[23];

  const size_t M = M_ROWSC;
  char* ws = (char*)d_ws;
  size_t off = 0;
  auto carve = [&](size_t bytes) -> void* {
    void* p = ws + off;
    off = (off + bytes + 255) & ~(size_t)255;
    return p;
  };
  float*  hidden   = (float*) carve(M * D_MODELC * 4);
  float*  residual = (float*) carve(M * D_MODELC * 4);
  __bf16* normed   = (__bf16*)carve(M * D_MODELC * 2);
  float*  xz       = (float*) carve(M * 2 * D_INNERC * 4);
  float*  xc       = (float*) carve(M * D_INNERC * 4);
  __bf16* xc_bf    = (__bf16*)carve(M * D_INNERC * 2);
  float*  dbl      = (float*) carve(M * XP_PADC * 4);
  float*  dtb      = (float*) carve(M * D_INNERC * 4);
  __bf16* y_bf     = (__bf16*)carve(M * D_INNERC * 2);
  __bf16* w_in_bf  = (__bf16*)carve((size_t)2 * D_INNERC * D_MODELC * 2);
  __bf16* w_xp_bf  = (__bf16*)carve((size_t)XP_PADC * D_INNERC * 2);
  __bf16* w_out_bf = (__bf16*)carve((size_t)D_MODELC * D_INNERC * 2);
  float*  v_sem    = (float*) carve((size_t)BATCHC * D_MODELC * 4);
  float*  v_stat   = (float*) carve((size_t)BATCHC * 128 * 4);

  auto cdiv = [](int a, int b) { return (a + b - 1) / b; };

  // 1) patch embed
  {
    int total = BATCHC * L_SEQC * D_MODELC;
    patch_embed_kernel<<<cdiv(total, 256), 256, 0, stream>>>(imgs, pe_w, pe_b, cls, pos, hidden);
  }

  // 2) mamba layers
  for (int i = 0; i < DEPTHC; ++i) {
    rms_norm_kernel<<<(int)M, 32, 0, stream>>>(hidden, residual,
                                               norm_ws + i * D_MODELC, normed, i == 0 ? 1 : 0);

    int nin = 2 * D_INNERC * D_MODELC;
    cvt_bf16_kernel<<<cdiv(nin, 256), 256, 0, stream>>>(in_proj_ws + (size_t)i * nin, w_in_bf, nin);

    {   // in_proj: [M,192] x [768,192]^T -> xz [M,768]
      int waves = ((int)M / 32) * ((2 * D_INNERC) / 32);
      wmma_gemm_bf16_kernel<<<cdiv(waves, 8), 256, 0, stream>>>(normed, w_in_bf, xz,
                                                                (int)M, 2 * D_INNERC, D_MODELC);
    }

    conv_silu_kernel<<<cdiv((int)M * D_INNERC, 256), 256, 0, stream>>>(
        xz, conv_ws + (size_t)i * D_INNERC * D_CONVC, conv_bs + (size_t)i * D_INNERC, xc, xc_bf);

    cvt_xp_pad_kernel<<<cdiv(XP_PADC * D_INNERC, 256), 256, 0, stream>>>(
        x_proj_ws + (size_t)i * (DT_RANKC + 2 * D_STATEC) * D_INNERC, w_xp_bf);

    {   // x_proj: [M,384] x [64,384]^T -> dbl [M,64]
      int waves = ((int)M / 32) * (XP_PADC / 32);
      wmma_gemm_bf16_kernel<<<cdiv(waves, 8), 256, 0, stream>>>(xc_bf, w_xp_bf, dbl,
                                                                (int)M, XP_PADC, D_INNERC);
    }

    dt_kernel<<<cdiv((int)M * D_INNERC, 256), 256, 0, stream>>>(
        dbl, dt_proj_ws + (size_t)i * D_INNERC * DT_RANKC,
        dt_proj_bs + (size_t)i * D_INNERC, dtb);

    scan_kernel<<<BATCHC, D_INNERC, 0, stream>>>(
        dbl, dtb, xc, xz, A_logs + (size_t)i * D_INNERC * D_STATEC,
        Ds + (size_t)i * D_INNERC, y_bf);

    int nout = D_MODELC * D_INNERC;
    cvt_bf16_kernel<<<cdiv(nout, 256), 256, 0, stream>>>(out_proj_ws + (size_t)i * nout, w_out_bf, nout);

    {   // out_proj: [M,384] x [192,384]^T -> hidden [M,192]
      int waves = ((int)M / 32) * (D_MODELC / 32);
      wmma_gemm_bf16_kernel<<<cdiv(waves, 8), 256, 0, stream>>>(y_bf, w_out_bf, hidden,
                                                                (int)M, D_MODELC, D_INNERC);
    }
  }

  // 3) final norm (cls row only) + stat MLP + head
  final_vsem_kernel<<<BATCHC, 32, 0, stream>>>(hidden, residual, norm_f_w, v_sem);
  stat_mlp_kernel<<<BATCHC, 128, 0, stream>>>(pl, iat, mlp1_w, mlp1_b, mlp2_w, mlp2_b, v_stat);
  head_kernel<<<BATCHC, 32, 0, stream>>>(v_sem, v_stat, head_w, head_b, (float*)d_out);
}
